// Encoder_90460601189274
// MI455X (gfx1250) — matrix operand order repaired and checked by reading, not compiled, not run on hardware
//
#include <hip/hip_runtime.h>
#include <hip/hip_bf16.h>

// ---------------------------------------------------------------------------
// GCN encoder for MI455X (gfx1250, wave32, WMMA + Tensor Data Mover)
//   featb = bf16(feat); w1t = bf16(W1^T)
//   h  = feat @ W1                (bf16 WMMA, TDM double-buffered LDS tiles)
//   s  = scatter-add over edges   (global_atomic_add_f32)
//   z  = bf16( BN(relu(s)) )      (batch stats via per-channel atomics)
//   t  = z @ W2                   (bf16 WMMA, W2 resident in LDS)
//   out= scatter-add over edges
// ---------------------------------------------------------------------------

typedef __attribute__((ext_vector_type(16))) __bf16   v16bf;
typedef __attribute__((ext_vector_type(8)))  float    v8f;
typedef __attribute__((ext_vector_type(8)))  unsigned uv8;
typedef __attribute__((ext_vector_type(4)))  unsigned uv4;
typedef __attribute__((ext_vector_type(8)))  int      iv8;
typedef __attribute__((ext_vector_type(4)))  int      iv4;

#define IN_FEAT  1024
#define HIDDEN   256
#define OUT_FEAT 64
#define BN_EPS   1e-5f

#define KCHUNK   64            // K elems per TDM chunk
#define NCHUNK   (IN_FEAT / KCHUNK)
#define LDSTR    72            // LDS row stride in bf16 elems (64 + 8 pad = 144B)

static __device__ __forceinline__ unsigned short f2bf(float f) {
  unsigned u = __builtin_bit_cast(unsigned, f);
  u += 0x7FFFu + ((u >> 16) & 1u);       // round-to-nearest-even
  return (unsigned short)(u >> 16);
}
static __device__ __forceinline__ unsigned pack2(float lo, float hi) {
  return (unsigned)f2bf(lo) | ((unsigned)f2bf(hi) << 16);
}

static __device__ __forceinline__ v8f wmma_bf16(v16bf a, v16bf b, v8f c) {
  return __builtin_amdgcn_wmma_f32_16x16x32_bf16(
      false, a, false, b, (short)0, c, false, false);
}

// A 16x32 bf16 fragment: lane L -> row M=L%16, khalf=L/16.
// VGPR v<4: K=2v+8*kh ; v>=4: K=16+2(v-4)+8*kh; K-pairs packed per dword.
static __device__ __forceinline__ v16bf load_fragA(const unsigned short* base, int kh) {
  uv8 u;
#pragma unroll
  for (int v = 0; v < 8; ++v) {
    int kb = ((v & 4) << 2) + ((v & 3) << 1) + (kh << 3);
    u[v] = *(const unsigned*)(base + kb);
  }
  return __builtin_bit_cast(v16bf, u);
}
// B 32x16 bf16 fragment: lane L -> col N=L%16, khalf=L/16; VGPR v<4: K=2v+8*kh ; v>=4: K=16+2(v-4)+8*kh.
static __device__ __forceinline__ v16bf load_fragB(const unsigned short* base, int kh) {
  uv8 u;
#pragma unroll
  for (int v = 0; v < 8; ++v) {
    int kb = ((v & 4) << 2) + ((v & 3) << 1) + (kh << 3);
    u[v] = *(const unsigned*)(base + kb);
  }
  return __builtin_bit_cast(v16bf, u);
}

// ---------------------------------------------------------------------------
// Tensor Data Mover: 2D tile (tile_d0 contiguous elems x tile_d1 rows) of bf16
// from global to LDS, padding each 128B row by 16B -> LDS stride 144B (72 el).
// D# bit layout per CDNA5 ISA ch.8 (group0: count/lds/global/type,
// group1: flags+dims+stride; groups 2,3 zero for 2D tiles).
// ---------------------------------------------------------------------------
static __device__ __forceinline__ void tdm_load_tile_bf16(
    unsigned lds_addr, unsigned long long gaddr,
    unsigned tensor_d0, unsigned tensor_d1, unsigned long long stride0,
    unsigned tile_d0, unsigned tile_d1)
{
  uv4 g0;
  g0[0] = 1u;                                            // count=1, user mode
  g0[1] = lds_addr;                                      // LDS byte address
  g0[2] = (unsigned)gaddr;                               // global addr [31:0]
  g0[3] = (unsigned)((gaddr >> 32) & 0x1FFFFFFull)       // global addr [56:32]
        | (2u << 30);                                    // type = 2 ("image")
  iv8 g1;
  g1[0] = (int)((1u << 16)        // data_size = 1 (2 bytes)
        |       (1u << 20)        // pad_enable
        |       (4u << 22)        // pad_interval: every 32 DWORDs (128B)
        |       (3u << 25));      // pad_amount: 4 DWORDs (16B)
  g1[1] = (int)((tensor_d0 & 0xFFFFu) << 16);            // dim0[15:0] @ bit48
  g1[2] = (int)((tensor_d0 >> 16) | ((tensor_d1 & 0xFFFFu) << 16));
  g1[3] = (int)((tensor_d1 >> 16) | (tile_d0 << 16));    // tile_dim0 @ bit112
  g1[4] = (int)(tile_d1 & 0xFFFFu);                      // tile_dim1; tile_dim2=0
  g1[5] = (int)(unsigned)stride0;                        // dim0_stride[31:0]
  g1[6] = (int)((unsigned)(stride0 >> 32) & 0xFFFFu);    // dim0_stride[47:32]
  g1[7] = 0;
  iv4 gz = {0, 0, 0, 0};
#if defined(__clang_major__) && (__clang_major__ >= 23)
  iv8 gz8 = {0, 0, 0, 0, 0, 0, 0, 0};
  __builtin_amdgcn_tensor_load_to_lds(g0, g1, gz, gz, gz8, 0);
#else
  __builtin_amdgcn_tensor_load_to_lds(g0, g1, gz, gz, 0);
#endif
}

// ---------------------------------------------------------------------------
// GEMM1: h[N,256] = featb[N,1024](bf16) @ w1t[256,1024](bf16, [n][k]).
// Block: 256 thr (8 waves), tile 128(M) x 64(N); wave = 32x32 (2x2 WMMA).
// Double-buffered TDM staging, one barrier per 64-wide K chunk.
// ---------------------------------------------------------------------------
__global__ __launch_bounds__(256) void gemm1_wmma(
    const unsigned short* __restrict__ featb, const unsigned short* __restrict__ w1t,
    float* __restrict__ h, int N)
{
  __shared__ unsigned short As[2][128][LDSTR];   // 2 x 18KB
  __shared__ unsigned short Bs[2][64][LDSTR];    // 2 x 9KB

  const int tid  = threadIdx.x;
  const int lane = tid & 31;
  const int w    = tid >> 5;
  const int lm   = lane & 15;
  const int kh   = lane >> 4;

  const int m0    = blockIdx.x * 128;
  const int n0    = blockIdx.y * 64;
  const int waveM = (w >> 1) * 32;
  const int waveN = (w & 1) * 32;
  const unsigned rows = (unsigned)min(128, N - m0);   // tail clamp for TDM

  v8f acc00 = {}, acc01 = {}, acc10 = {}, acc11 = {};

  if (w == 0) {       // chunk 0 into buffer 0
    tdm_load_tile_bf16((unsigned)(size_t)&As[0][0][0],
        (unsigned long long)(size_t)(featb + (size_t)m0 * IN_FEAT),
        IN_FEAT, (unsigned)N, IN_FEAT, KCHUNK, rows);
    tdm_load_tile_bf16((unsigned)(size_t)&Bs[0][0][0],
        (unsigned long long)(size_t)(w1t + (size_t)n0 * IN_FEAT),
        IN_FEAT, HIDDEN, IN_FEAT, KCHUNK, 64);
  }

  for (int c = 0; c < NCHUNK; ++c) {
    if (w == 0) __builtin_amdgcn_s_wait_tensorcnt(0);   // chunk c landed
    __syncthreads();   // c visible to all; all waves done reading buf (c+1)&1
    if (w == 0 && (c + 1) < NCHUNK) {
      const int k1 = (c + 1) * KCHUNK, b1 = (c + 1) & 1;
      tdm_load_tile_bf16((unsigned)(size_t)&As[b1][0][0],
          (unsigned long long)(size_t)(featb + (size_t)m0 * IN_FEAT + k1),
          IN_FEAT, (unsigned)N, IN_FEAT, KCHUNK, rows);
      tdm_load_tile_bf16((unsigned)(size_t)&Bs[b1][0][0],
          (unsigned long long)(size_t)(w1t + (size_t)n0 * IN_FEAT + k1),
          IN_FEAT, HIDDEN, IN_FEAT, KCHUNK, 64);
    }
    const int buf = c & 1;
#pragma unroll
    for (int s = 0; s < 2; ++s) {      // two 16x16x32 K-steps per chunk
      v16bf a0 = load_fragA(&As[buf][waveM +  0 + lm][s * 32], kh);
      v16bf a1 = load_fragA(&As[buf][waveM + 16 + lm][s * 32], kh);
      v16bf b0 = load_fragB(&Bs[buf][waveN +  0 + lm][s * 32], kh);
      v16bf b1 = load_fragB(&Bs[buf][waveN + 16 + lm][s * 32], kh);
      acc00 = wmma_bf16(a0, b0, acc00);
      acc01 = wmma_bf16(a0, b1, acc01);
      acc10 = wmma_bf16(a1, b0, acc10);
      acc11 = wmma_bf16(a1, b1, acc11);
    }
  }

  // C/D layout: lane col = lm; VGPR v -> row v + 8*khalf
  const int nOut = n0 + waveN + lm;
#pragma unroll
  for (int v = 0; v < 8; ++v) {
    int m = m0 + waveM + v + (kh << 3);
    if (m < N) {
      h[(size_t)m * HIDDEN + nOut]      = acc00[v];
      h[(size_t)m * HIDDEN + nOut + 16] = acc01[v];
    }
    int m2 = m + 16;
    if (m2 < N) {
      h[(size_t)m2 * HIDDEN + nOut]      = acc10[v];
      h[(size_t)m2 * HIDDEN + nOut + 16] = acc11[v];
    }
  }
}

// ---------------------------------------------------------------------------
// GEMM2: t[N,64] = z[N,256](bf16) @ W2bf[256,64]; W2 fully LDS-resident.
// ---------------------------------------------------------------------------
__global__ __launch_bounds__(256) void gemm2_wmma(
    const unsigned short* __restrict__ z, const unsigned short* __restrict__ w2b,
    float* __restrict__ t, int N)
{
  __shared__ unsigned short Bs[64][264];

  const int tid  = threadIdx.x;
  const int lane = tid & 31;
  const int w    = tid >> 5;
  const int lm   = lane & 15;
  const int kh   = lane >> 4;

  for (int i = tid; i < HIDDEN * OUT_FEAT; i += 256) {
    int k = i >> 6, n = i & 63;
    Bs[n][k] = w2b[i];
  }
  __syncthreads();

  const int m0    = blockIdx.x * 32;
  const int waveM = (w >> 2) * 16;
  const int waveN = (w & 3) * 16;

  const int m  = m0 + waveM + lm;
  const int mc = (m < N) ? m : (N - 1);           // clamp; stores are guarded
  const unsigned short* zrow = z + (size_t)mc * HIDDEN;
  const unsigned short* bcol = &Bs[waveN + lm][0];

  v8f acc = {};
#pragma unroll
  for (int k0 = 0; k0 < HIDDEN; k0 += 32) {
    uv8 au, bu;
#pragma unroll
    for (int v = 0; v < 8; ++v) {
      int kbA = ((v & 4) << 2) + ((v & 3) << 1) + (kh << 3);
      au[v] = *(const unsigned*)(zrow + k0 + kbA);
      int kbB = ((v & 4) << 2) + ((v & 3) << 1) + (kh << 3);
      bu[v] = *(const unsigned*)(bcol + k0 + kbB);
    }
    acc = wmma_bf16(__builtin_bit_cast(v16bf, au),
                    __builtin_bit_cast(v16bf, bu), acc);
  }

  const int nOut = waveN + lm;
#pragma unroll
  for (int v = 0; v < 8; ++v) {
    int mm = m0 + waveM + v + (kh << 3);
    if (mm < N) t[(size_t)mm * OUT_FEAT + nOut] = acc[v];
  }
}

// ---------------------------------------------------------------------------
// SpMM scatter: out[row] += val * x[col]; one wave32 per edge.
// ---------------------------------------------------------------------------
__global__ void spmm_scatter256(const int* __restrict__ row, const int* __restrict__ col,
                                const float* __restrict__ val, const float* __restrict__ x,
                                float* __restrict__ out, int E)
{
  int gid = blockIdx.x * blockDim.x + threadIdx.x;
  int e = gid >> 5;
  if (e >= E) return;
  int lane = gid & 31;
  int r = row[e], c = col[e];
  float v = val[e];
  const float* src = x + (size_t)c * HIDDEN;
  float* dst       = out + (size_t)r * HIDDEN;
#pragma unroll
  for (int j = 0; j < 8; ++j) {
    int ch = lane + j * 32;                 // coalesced per wave
    atomicAdd(dst + ch, v * src[ch]);
  }
}

__global__ void spmm_scatter64(const int* __restrict__ row, const int* __restrict__ col,
                               const float* __restrict__ val, const float* __restrict__ x,
                               float* __restrict__ out, int E)
{
  int gid = blockIdx.x * blockDim.x + threadIdx.x;
  int e = gid >> 5;
  if (e >= E) return;
  int lane = gid & 31;
  int r = row[e], c = col[e];
  float v = val[e];
  const float* src = x + (size_t)c * OUT_FEAT;
  float* dst       = out + (size_t)r * OUT_FEAT;
  atomicAdd(dst + lane,      v * src[lane]);
  atomicAdd(dst + lane + 32, v * src[lane + 32]);
}

// ---------------------------------------------------------------------------
// ReLU + BatchNorm (training stats)
// ---------------------------------------------------------------------------
__global__ __launch_bounds__(256) void bn_stats(const float* __restrict__ s,
                                                float* __restrict__ stats, int N)
{
  int c  = threadIdx.x;
  int r0 = blockIdx.x * 128;
  int r1 = min(r0 + 128, N);
  float sm = 0.f, sq = 0.f;
  for (int r = r0; r < r1; ++r) {
    float x = s[(size_t)r * HIDDEN + c];
    x = fmaxf(x, 0.f);
    sm += x; sq += x * x;
  }
  atomicAdd(stats + c, sm);
  atomicAdd(stats + HIDDEN + c, sq);
}

__global__ void bn_final(float* __restrict__ stats, const float* __restrict__ gamma,
                         const float* __restrict__ beta, int N)
{
  int c = threadIdx.x;
  float inv  = 1.f / (float)N;
  float mean = stats[c] * inv;
  float var  = stats[HIDDEN + c] * inv - mean * mean;
  float sc   = gamma[c] * rsqrtf(var + BN_EPS);
  stats[2 * HIDDEN + c] = sc;
  stats[3 * HIDDEN + c] = beta[c] - mean * sc;
}

__global__ void bn_apply(const float* __restrict__ s, const float* __restrict__ stats,
                         unsigned short* __restrict__ z, size_t n)
{
  size_t i = (size_t)blockIdx.x * blockDim.x + threadIdx.x;
  if (i >= n) return;
  int c = (int)(i & (HIDDEN - 1));
  float x = fmaxf(s[i], 0.f);
  z[i] = f2bf(x * stats[2 * HIDDEN + c] + stats[3 * HIDDEN + c]);
}

// ---------------------------------------------------------------------------
// Conversions / utility
// ---------------------------------------------------------------------------
__global__ void f32_to_bf16_v4(const float* __restrict__ src,
                               unsigned short* __restrict__ dst, size_t n4)
{
  size_t i = (size_t)blockIdx.x * blockDim.x + threadIdx.x;
  if (i >= n4) return;
  float4 f = ((const float4*)src)[i];
  ((uint2*)dst)[i] = make_uint2(pack2(f.x, f.y), pack2(f.z, f.w));
}

__global__ void w1_transpose_bf16(const float* __restrict__ W1,
                                  unsigned short* __restrict__ w1t)
{
  int i = blockIdx.x * blockDim.x + threadIdx.x;     // over 256*1024
  if (i >= HIDDEN * IN_FEAT) return;
  int n = i >> 10, k = i & (IN_FEAT - 1);
  w1t[i] = f2bf(W1[(size_t)k * HIDDEN + n]);
}

__global__ void w2_convert(const float* __restrict__ W2,
                           unsigned short* __restrict__ w2b, int n)
{
  int i = blockIdx.x * blockDim.x + threadIdx.x;
  if (i < n) w2b[i] = f2bf(W2[i]);
}

__global__ void zero_f32(float* __restrict__ p, size_t n)
{
  size_t i = (size_t)blockIdx.x * blockDim.x + threadIdx.x;
  size_t st = (size_t)gridDim.x * blockDim.x;
  for (; i < n; i += st) p[i] = 0.f;
}

// ---------------------------------------------------------------------------
extern "C" void kernel_launch(void* const* d_in, const int* in_sizes, int n_in,
                              void* d_out, int out_size, void* d_ws, size_t ws_size,
                              hipStream_t stream)
{
  const float* feat  = (const float*)d_in[0];
  const int*   erow  = (const int*)  d_in[1];
  const int*   ecol  = (const int*)  d_in[2];
  const float* eval  = (const float*)d_in[3];
  const float* W1    = (const float*)d_in[4];
  const float* W2    = (const float*)d_in[5];
  const float* gamma = (const float*)d_in[6];
  const float* beta  = (const float*)d_in[7];
  float* out = (float*)d_out;

  const int N = in_sizes[0] / IN_FEAT;
  const int E = in_sizes[1];

  // workspace layout (~333 MB for N=100000):
  //   h[N*256]f32 | regB: featb[N*1024]bf16  (later aliased by s[N*256]f32 +
  //   z[N*256]bf16) | t[N*64]f32 | stats[1024]f32 | w2b[16K]bf16 | w1t[256K]bf16
  float* h = (float*)d_ws;
  char* regB = (char*)(h + (size_t)N * HIDDEN);
  unsigned short* featb = (unsigned short*)regB;
  float* s = (float*)regB;                                        // after GEMM1
  unsigned short* z = (unsigned short*)(regB + (size_t)N * HIDDEN * 4);
  float* t = (float*)(regB + (size_t)N * IN_FEAT * 2);
  float* stats = t + (size_t)N * OUT_FEAT;
  unsigned short* w2b = (unsigned short*)(stats + 4 * HIDDEN);
  unsigned short* w1t = w2b + HIDDEN * OUT_FEAT;

  // precision conversions for the WMMA pipeline
  {
    size_t n4 = (size_t)N * IN_FEAT / 4;
    f32_to_bf16_v4<<<dim3((unsigned)((n4 + 255) / 256)), 256, 0, stream>>>(feat, featb, n4);
  }
  w1_transpose_bf16<<<dim3((HIDDEN * IN_FEAT + 255) / 256), 256, 0, stream>>>(W1, w1t);
  w2_convert<<<dim3((HIDDEN * OUT_FEAT + 255) / 256), 256, 0, stream>>>(W2, w2b, HIDDEN * OUT_FEAT);

  // zero accumulators that don't alias featb
  zero_f32<<<dim3(4), 256, 0, stream>>>(stats, (size_t)(4 * HIDDEN));
  {
    size_t no = (size_t)N * OUT_FEAT;
    zero_f32<<<dim3((unsigned)((no + 255) / 256)), 256, 0, stream>>>(out, no);
  }

  // layer 1 dense GEMM (TDM + WMMA)
  gemm1_wmma<<<dim3((N + 127) / 128, HIDDEN / 64), 256, 0, stream>>>(featb, w1t, h, N);

  // featb now dead -> its region becomes s
  {
    size_t ns = (size_t)N * HIDDEN;
    zero_f32<<<dim3((unsigned)((ns + 255) / 256)), 256, 0, stream>>>(s, ns);
  }

  // sparse aggregation 1
  {
    long long thr = (long long)E * 32;
    spmm_scatter256<<<dim3((unsigned)((thr + 255) / 256)), 256, 0, stream>>>(erow, ecol, eval, h, s, E);
  }

  // relu + batchnorm (training stats)
  bn_stats<<<dim3((N + 127) / 128), 256, 0, stream>>>(s, stats, N);
  bn_final<<<dim3(1), 256, 0, stream>>>(stats, gamma, beta, N);
  {
    size_t n = (size_t)N * HIDDEN;
    bn_apply<<<dim3((unsigned)((n + 255) / 256)), 256, 0, stream>>>(s, stats, z, n);
  }

  // layer 2 dense GEMM
  gemm2_wmma<<<dim3((N + 31) / 32), 256, 0, stream>>>(z, w2b, t, N);

  // sparse aggregation 2 -> final output
  {
    long long thr = (long long)E * 32;
    spmm_scatter64<<<dim3((unsigned)((thr + 255) / 256)), 256, 0, stream>>>(erow, ecol, eval, t, out, E);
  }
}
